// LSTMClassifier_44341242364634
// MI455X (gfx1250) — matrix operand order repaired and checked by reading, not compiled, not run on hardware
//
#include <hip/hip_runtime.h>
#include <hip/hip_bf16.h>

// Problem constants (match reference)
#define DIM   1024
#define BATCH 128
#define TSTEP 256
#define FOURD 4096

typedef __bf16 bf16_t;
typedef __attribute__((ext_vector_type(16))) __bf16 bf16x16;
typedef __attribute__((ext_vector_type(8)))  __bf16 bf16x8;
typedef __attribute__((ext_vector_type(8)))  float  f32x8;

union ABPack { bf16x16 v; bf16x8 h[2]; };

__device__ __forceinline__ bf16_t f2bf(float f) {
  unsigned int u = __builtin_bit_cast(unsigned int, f);
  unsigned int r = u + 0x7FFFu + ((u >> 16) & 1u);   // round-to-nearest-even
  unsigned short s = (unsigned short)(r >> 16);
  return __builtin_bit_cast(bf16_t, s);
}

__device__ __forceinline__ float sigmoidf(float x) {
  return 1.0f / (1.0f + __expf(-x));
}

// ---- One-time setup kernels -------------------------------------------------

__global__ __launch_bounds__(256) void convert_weights(
    const float* __restrict__ wih, const float* __restrict__ whh,
    bf16_t* __restrict__ wihb, bf16_t* __restrict__ whhb) {
  int i = blockIdx.x * 256 + threadIdx.x;     // 0 .. 4M-1
  wihb[i] = f2bf(wih[i]);
  whhb[i] = f2bf(whh[i]);
}

__global__ __launch_bounds__(256) void init_state(
    const float* __restrict__ h0, const float* __restrict__ c0,
    const float* __restrict__ bih, const float* __restrict__ bhh,
    float* __restrict__ h, float* __restrict__ c,
    bf16_t* __restrict__ hb, float* __restrict__ bsum) {
  int i = blockIdx.x * 256 + threadIdx.x;     // grid covers BATCH*DIM
  if (i < BATCH * DIM) {
    float hv = h0[i];
    h[i] = hv;
    c[i] = c0[i];
    hb[i] = f2bf(hv);
  }
  if (i < FOURD) bsum[i] = bih[i] + bhh[i];
}

__global__ __launch_bounds__(256) void convert_x(
    const float* __restrict__ inp, bf16_t* __restrict__ xb, int t) {
  int i = blockIdx.x * 256 + threadIdx.x;     // 0 .. BATCH*DIM-1
  int b = i >> 10, d = i & (DIM - 1);
  xb[i] = f2bf(inp[((size_t)b * TSTEP + t) * DIM + d]);
}

// ---- Recurrent-step kernels -------------------------------------------------

// gates[B,4D] = xb[B,D]*Wih^T + hb[B,D]*Whh^T + bsum.
// 512 strips of 16 rows x 64 cols. Each strip is computed by a WAVE PAIR:
// even wave does the x-pass, odd wave does the h-pass; partials combined via
// LDS (padded stride 33 to dodge 64-bank conflicts). Per wave: 4 independent
// accumulators (breaks the bf16 WMMA->WMMA RAW chain, ISA 7.12.1) and a
// 4x-amortized A operand with the VOP3P matrix_a_reuse hint.
// 4 strips/block -> 128 blocks, 8 waves each.
__global__ __launch_bounds__(256) void lstm_gates_wmma(
    const bf16_t* __restrict__ xb,   // [B, D]   bf16
    const bf16_t* __restrict__ hb,   // [B, D]   bf16
    const bf16_t* __restrict__ wih,  // [4D, D]  bf16 (row n holds all k)
    const bf16_t* __restrict__ whh,  // [4D, D]  bf16
    const float*  __restrict__ bsum, // [4D]
    float* __restrict__ gates)       // [B, 4D]
{
  __shared__ float part[4][32][33];            // [pair][lane][32 accum vals]

  const int lane  = threadIdx.x & 31;
  const int wave  = threadIdx.x >> 5;
  const int pair  = wave >> 1;                 // 0..3
  const int half  = wave & 1;                  // 0: x-pass, 1: h-pass
  const int strip = blockIdx.x * 4 + pair;     // 512 strips total
  const int mtile = strip >> 6;                // 0..7   (B = 128)
  const int n0    = (strip & 63) * 64;         // first of 64 columns

  const int mrow = mtile * 16 + (lane & 15);   // A row (batch index)
  const int ncol = (lane & 15);                // column within a 16-wide tile
  const int ah   = (lane >> 4) * 8;            // D: row offset 8*(lane>=16)
  const int bh   = (lane >> 4) * 16;           // A and B: K sub-offset, contiguous k = 16h + s

  f32x8 acc0 = {}, acc1 = {}, acc2 = {}, acc3 = {};

  const bf16_t* __restrict__ arow = (half ? hb : xb) + (size_t)mrow * DIM;
  const bf16_t* __restrict__ w    = half ? whh : wih;
  const bf16_t* __restrict__ br0  = w + (size_t)(n0 +  0 + ncol) * DIM + bh;
  const bf16_t* __restrict__ br1  = w + (size_t)(n0 + 16 + ncol) * DIM + bh;
  const bf16_t* __restrict__ br2  = w + (size_t)(n0 + 32 + ncol) * DIM + bh;
  const bf16_t* __restrict__ br3  = w + (size_t)(n0 + 48 + ncol) * DIM + bh;

#pragma unroll 2
  for (int kk = 0; kk < DIM; kk += 32) {
    ABPack a, w0, w1, w2, w3;
    a.h[0]  = *(const bf16x8*)(arow + kk + bh);
    a.h[1]  = *(const bf16x8*)(arow + kk + bh + 8);
    w0.h[0] = *(const bf16x8*)(br0 + kk);
    w0.h[1] = *(const bf16x8*)(br0 + kk + 8);
    w1.h[0] = *(const bf16x8*)(br1 + kk);
    w1.h[1] = *(const bf16x8*)(br1 + kk + 8);
    w2.h[0] = *(const bf16x8*)(br2 + kk);
    w2.h[1] = *(const bf16x8*)(br2 + kk + 8);
    w3.h[0] = *(const bf16x8*)(br3 + kk);
    w3.h[1] = *(const bf16x8*)(br3 + kk + 8);
    // Same A operand for all 4: set the reuse-A hint on the first three.
    acc0 = __builtin_amdgcn_wmma_f32_16x16x32_bf16(
               false, a.v, false, w0.v, (short)0, acc0, true,  false);
    acc1 = __builtin_amdgcn_wmma_f32_16x16x32_bf16(
               false, a.v, false, w1.v, (short)0, acc1, true,  false);
    acc2 = __builtin_amdgcn_wmma_f32_16x16x32_bf16(
               false, a.v, false, w2.v, (short)0, acc2, true,  false);
    acc3 = __builtin_amdgcn_wmma_f32_16x16x32_bf16(
               false, a.v, false, w3.v, (short)0, acc3, false, false);
  }

  if (half == 0) {   // x-pass wave parks its partial sums in LDS
#pragma unroll
    for (int v = 0; v < 8; ++v) {
      part[pair][lane][v     ] = acc0[v];
      part[pair][lane][v +  8] = acc1[v];
      part[pair][lane][v + 16] = acc2[v];
      part[pair][lane][v + 24] = acc3[v];
    }
  }
  __syncthreads();
  if (half == 1) {   // h-pass wave combines, adds bias, stores gates
    const int   rbase = mtile * 16 + ah;  // D-layout: VGPR v -> M = v + 8*(lane>=16)
    const float bias0 = bsum[n0 +  0 + ncol];
    const float bias1 = bsum[n0 + 16 + ncol];
    const float bias2 = bsum[n0 + 32 + ncol];
    const float bias3 = bsum[n0 + 48 + ncol];
#pragma unroll
    for (int v = 0; v < 8; ++v) {
      float* grow = gates + (size_t)(rbase + v) * FOURD + n0 + ncol;
      grow[ 0] = acc0[v] + part[pair][lane][v     ] + bias0;
      grow[16] = acc1[v] + part[pair][lane][v +  8] + bias1;
      grow[32] = acc2[v] + part[pair][lane][v + 16] + bias2;
      grow[48] = acc3[v] + part[pair][lane][v + 24] + bias3;
    }
  }
}

// Gate activations + state update, fused with the bf16 conversion of the
// NEXT step's x slice (independent of this step's results).
__global__ __launch_bounds__(256) void lstm_pointwise(
    const float* __restrict__ gates,
    float* __restrict__ c, float* __restrict__ h, bf16_t* __restrict__ hb,
    const float* __restrict__ inp, bf16_t* __restrict__ xb, int tnext) {
  int i = blockIdx.x * 256 + threadIdx.x;     // 0 .. BATCH*DIM-1
  int b = i >> 10, d = i & (DIM - 1);
  const float* g = gates + (size_t)b * FOURD;
  float ig = sigmoidf(g[d]);
  float fg = sigmoidf(g[d + 1024]);
  float gg = tanhf(g[d + 2048]);
  float og = sigmoidf(g[d + 3072]);
  float cn = fg * c[i] + ig * gg;
  float hn = og * tanhf(cn);
  c[i] = cn;
  h[i] = hn;
  hb[i] = f2bf(hn);
  if (tnext < TSTEP)
    xb[i] = f2bf(inp[((size_t)b * TSTEP + tnext) * DIM + d]);
}

// ---- Final head -------------------------------------------------------------

// yhat[b] = sigmoid(dot(h[b,:], fc_w) + fc_b). One block per batch row.
__global__ __launch_bounds__(256) void final_fc(
    const float* __restrict__ h, const float* __restrict__ fcw,
    const float* __restrict__ fcb, float* __restrict__ out) {
  __shared__ float red[256];
  int b = blockIdx.x;
  float p = 0.0f;
  for (int k = threadIdx.x; k < DIM; k += 256)
    p += h[(size_t)b * DIM + k] * fcw[k];
  red[threadIdx.x] = p;
  __syncthreads();
  for (int s = 128; s > 0; s >>= 1) {
    if (threadIdx.x < s) red[threadIdx.x] += red[threadIdx.x + s];
    __syncthreads();
  }
  if (threadIdx.x == 0) out[b] = sigmoidf(red[0] + fcb[0]);
}

__global__ __launch_bounds__(256) void copy_hc(
    const float* __restrict__ h, const float* __restrict__ c,
    float* __restrict__ out) {
  int i = blockIdx.x * 256 + threadIdx.x;     // 0 .. BATCH*DIM-1
  out[BATCH + i] = h[i];
  out[BATCH + BATCH * DIM + i] = c[i];
}

// ---- Launch -----------------------------------------------------------------

extern "C" void kernel_launch(void* const* d_in, const int* in_sizes, int n_in,
                              void* d_out, int out_size, void* d_ws, size_t ws_size,
                              hipStream_t stream) {
  const float* inp  = (const float*)d_in[0];   // [B, T, D]
  const float* h0   = (const float*)d_in[1];   // [B, D]
  const float* c0   = (const float*)d_in[2];   // [B, D]
  const float* W_ih = (const float*)d_in[3];   // [4D, D]
  const float* W_hh = (const float*)d_in[4];   // [4D, D]
  const float* b_ih = (const float*)d_in[5];   // [4D]
  const float* b_hh = (const float*)d_in[6];   // [4D]
  const float* fc_w = (const float*)d_in[7];   // [1, D]
  const float* fc_b = (const float*)d_in[8];   // [1]

  // Workspace layout (256-byte aligned slabs), ~20.5 MB total
  char* ws = (char*)d_ws;
  size_t off = 0;
  auto take = [&](size_t bytes) {
    char* p = ws + off;
    off = (off + bytes + 255) & ~(size_t)255;
    return p;
  };
  bf16_t* wihb = (bf16_t*)take((size_t)FOURD * DIM * sizeof(bf16_t));  // 8 MB
  bf16_t* whhb = (bf16_t*)take((size_t)FOURD * DIM * sizeof(bf16_t));  // 8 MB
  float*  bsum = (float*) take((size_t)FOURD * sizeof(float));
  float*  h    = (float*) take((size_t)BATCH * DIM * sizeof(float));
  float*  c    = (float*) take((size_t)BATCH * DIM * sizeof(float));
  bf16_t* hb   = (bf16_t*)take((size_t)BATCH * DIM * sizeof(bf16_t));
  bf16_t* xb   = (bf16_t*)take((size_t)BATCH * DIM * sizeof(bf16_t));
  float*  gate = (float*) take((size_t)BATCH * FOURD * sizeof(float)); // 2 MB

  float* out = (float*)d_out;

  // One-time setup
  convert_weights<<<(FOURD * DIM) / 256, 256, 0, stream>>>(W_ih, W_hh, wihb, whhb);
  init_state<<<(BATCH * DIM) / 256, 256, 0, stream>>>(h0, c0, b_ih, b_hh, h, c, hb, bsum);
  convert_x<<<(BATCH * DIM) / 256, 256, 0, stream>>>(inp, xb, 0);

  // Recurrent loop: 256 sequential steps, 2 launches each
  for (int t = 0; t < TSTEP; ++t) {
    lstm_gates_wmma<<<128, 256, 0, stream>>>(xb, hb, wihb, whhb, bsum, gate);
    lstm_pointwise<<<(BATCH * DIM) / 256, 256, 0, stream>>>(
        gate, c, h, hb, inp, xb, t + 1);
  }

  // Head + output tuple (yhat[128], hx[128*1024], cx[128*1024])
  final_fc<<<BATCH, 256, 0, stream>>>(h, fc_w, fc_b, out);
  copy_hc<<<(BATCH * DIM) / 256, 256, 0, stream>>>(h, c, out);
}
